// Attention_31568009626346
// MI455X (gfx1250) — matrix operand order repaired and checked by reading, not compiled, not run on hardware
//
#include <hip/hip_runtime.h>

#define DEVFN __device__ __forceinline__

typedef __bf16 bf16;
typedef __attribute__((ext_vector_type(8)))  bf16  v8bf;
typedef __attribute__((ext_vector_type(16))) bf16  v16bf;
typedef __attribute__((ext_vector_type(8)))  float v8f;
typedef __attribute__((ext_vector_type(4)))  unsigned int u32x4;
typedef __attribute__((ext_vector_type(8)))  int i32x8;
typedef __attribute__((ext_vector_type(4)))  int i32x4;

constexpr int Bsz  = 2;
constexpr int Seq  = 2048;
constexpr int Hid  = 1024;
constexpr int NHd  = 16;
constexpr int HDm  = 64;
constexpr int Mtok = Bsz * Seq;   // 4096

union F16x16 { v16bf v; v8bf h[2]; };

// round-to-nearest-even f32 -> bf16 bits
DEVFN unsigned bfbits(float f) {
  unsigned u = __float_as_uint(f);
  return (u + 0x7FFFu + ((u >> 16) & 1u)) >> 16;
}

// A-matrix fragment, 16(M) x 32(K) bf16, row-major source with leading dim ld.
// lanes 0-15 -> M=lane, K={0..7,16..23}; lanes 16-31 -> M=lane-16, K={8..15,24..31}
DEVFN v16bf load_fragA(const unsigned short* p0, int ld) {
  const int lane = threadIdx.x & 31;
  const int row  = lane & 15;
  const int k0   = (lane & 16) ? 8 : 0;
  const bf16* p  = (const bf16*)p0 + (size_t)row * ld + k0;
  F16x16 f;
  f.h[0] = *(const v8bf*)(p);
  f.h[1] = *(const v8bf*)(p + 16);
  return f.v;
}

// B-matrix fragment, 32(K) x 16(N) bf16; source row-major over K with rows = N
// (B[k][n] = p0[n*ld + k]).  lane = N; lanes 0-15 hold K={0..7,16..23}, lanes 16-31 K={8..15,24..31}.
DEVFN v16bf load_fragB(const unsigned short* p0, int ld) {
  const int lane = threadIdx.x & 31;
  const int n    = lane & 15;
  const int k0   = (lane & 16) ? 8 : 0;
  const bf16* p  = (const bf16*)p0 + (size_t)n * ld + k0;
  F16x16 f;
  f.h[0] = *(const v8bf*)(p);
  f.h[1] = *(const v8bf*)(p + 16);
  return f.v;
}

DEVFN v8f wmma_bf16(v16bf a, v16bf b, v8f c) {
  return __builtin_amdgcn_wmma_f32_16x16x32_bf16(
      false, a, false, b, (short)0, c, false, false);
}

// ---------------------------------------------------------------------------
// Tensor Data Mover: 2D tile (bf16) from global -> LDS.
//   tile_d0 = contiguous elements per row, tile_d1 = rows,
//   stride0 = row stride in elements.  LDS receives the tile packed
//   row-major (ld = tile_d0).  D# per CDNA5 ISA 8.3/8.4.
// ---------------------------------------------------------------------------
DEVFN unsigned lds_off(const void* p) { return (unsigned)(size_t)p; }

DEVFN void tdm_load_2d(unsigned ldsAddr, const unsigned short* gsrc,
                       unsigned tile_d0, unsigned tile_d1,
                       unsigned tensor_d0, unsigned tensor_d1,
                       unsigned long long stride0) {
  unsigned long long ga = (unsigned long long)(size_t)gsrc;
  u32x4 g0;
  g0[0] = 1u;                                   // count=1, user descriptor
  g0[1] = ldsAddr;                              // lds_addr (bytes)
  g0[2] = (unsigned)(ga & 0xFFFFFFFFu);         // global_addr[31:0]
  g0[3] = (unsigned)((ga >> 32) & 0x01FFFFFFu)  // global_addr[56:32]
        | (2u << 30);                           // type = 2 ("image")
  i32x8 g1;
  g1[0] = (int)(1u << 16);                                  // data_size=2B
  g1[1] = (int)((tensor_d0 & 0xFFFFu) << 16);               // tensor_dim0 lo
  g1[2] = (int)((tensor_d0 >> 16) | ((tensor_d1 & 0xFFFFu) << 16));
  g1[3] = (int)((tensor_d1 >> 16) | (tile_d0 << 16));       // tile_dim0
  g1[4] = (int)(tile_d1 & 0xFFFFu);                         // tile_dim1 (dim2=0)
  g1[5] = (int)(unsigned)(stride0 & 0xFFFFFFFFu);           // dim0 stride lo
  g1[6] = (int)(unsigned)((stride0 >> 32) & 0xFFFFu);       // dim0 stride hi
  g1[7] = 0;
  const i32x4 z4 = {0, 0, 0, 0};
#if defined(__clang_major__) && (__clang_major__ >= 23)
  const i32x8 z8 = {0, 0, 0, 0, 0, 0, 0, 0};
  __builtin_amdgcn_tensor_load_to_lds(g0, g1, z4, z4, z8, 0);
#else
  __builtin_amdgcn_tensor_load_to_lds(g0, g1, z4, z4, 0);
#endif
}

// ---------------------------------------------------------------------------
// fp32 -> bf16 bulk conversion (4 elems/thread)
// ---------------------------------------------------------------------------
__global__ __launch_bounds__(256) void f32_to_bf16_kernel(
    const float* __restrict__ src, unsigned short* __restrict__ dst, int n4) {
  int i = blockIdx.x * 256 + threadIdx.x;
  if (i >= n4) return;
  const float4 f = ((const float4*)src)[i];
  uint2 o;
  o.x = bfbits(f.x) | (bfbits(f.y) << 16);
  o.y = bfbits(f.z) | (bfbits(f.w) << 16);
  ((uint2*)dst)[i] = o;
}

// ---------------------------------------------------------------------------
// Y = X @ W^T + bias, bf16 out in head-split layout.  Register-double-buffered
// K loop: fragments for k+32 are in flight while WMMAs for k execute.
// Block: 256 thr = 8 waves (2x4); wave tile 32x32; block tile 64x128.
// ---------------------------------------------------------------------------
__global__ __launch_bounds__(256) void qkv_gemm_kernel(
    const unsigned short* __restrict__ X,
    const unsigned short* __restrict__ W,
    const float* __restrict__ bias,
    unsigned short* __restrict__ dst,
    int transposed) {
  const int wave = threadIdx.x >> 5;
  const int m = blockIdx.x * 64  + (wave >> 2) * 32;
  const int n = blockIdx.y * 128 + (wave & 3) * 32;

  const unsigned short* xA0 = X + (size_t)m        * Hid;
  const unsigned short* xA1 = X + (size_t)(m + 16) * Hid;
  const unsigned short* wB0 = W + (size_t)n        * Hid;
  const unsigned short* wB1 = W + (size_t)(n + 16) * Hid;

  v16bf a0 = load_fragA(xA0, Hid);
  v16bf a1 = load_fragA(xA1, Hid);
  v16bf b0 = load_fragB(wB0, Hid);
  v16bf b1 = load_fragB(wB1, Hid);

  v8f acc00 = {}, acc01 = {}, acc10 = {}, acc11 = {};

  for (int k = 32; k < Hid; k += 32) {
    const v16bf na0 = load_fragA(xA0 + k, Hid);
    const v16bf na1 = load_fragA(xA1 + k, Hid);
    const v16bf nb0 = load_fragB(wB0 + k, Hid);
    const v16bf nb1 = load_fragB(wB1 + k, Hid);
    if (k + 32 < Hid) {
      __builtin_prefetch(xA0 + k + 32, 0, 1);
      __builtin_prefetch(wB0 + k + 32, 0, 1);
    }
    acc00 = wmma_bf16(a0, b0, acc00);
    acc01 = wmma_bf16(a0, b1, acc01);
    acc10 = wmma_bf16(a1, b0, acc10);
    acc11 = wmma_bf16(a1, b1, acc11);
    a0 = na0; a1 = na1; b0 = nb0; b1 = nb1;
  }
  acc00 = wmma_bf16(a0, b0, acc00);
  acc01 = wmma_bf16(a0, b1, acc01);
  acc10 = wmma_bf16(a1, b0, acc10);
  acc11 = wmma_bf16(a1, b1, acc11);

  const int lane = threadIdx.x & 31;
  const int nn   = lane & 15;
  const int half = lane >> 4;
  const v8f* accs[4] = {&acc00, &acc01, &acc10, &acc11};
#pragma unroll
  for (int im = 0; im < 2; im++) {
#pragma unroll
    for (int in = 0; in < 2; in++) {
      const v8f& a = *accs[im * 2 + in];
#pragma unroll
      for (int r = 0; r < 8; r++) {
        const int row = m + im * 16 + half * 8 + r;   // token index
        const int col = n + in * 16 + nn;             // h*HD + d
        const float val = a[r] + bias[col];
        const int bb = row >> 11;             // / Seq
        const int ss = row & (Seq - 1);
        const int hh = col >> 6;              // / HDm
        const int dd = col & (HDm - 1);
        const size_t idx = transposed
            ? ((size_t)(bb * NHd + hh) * HDm + dd) * Seq + ss
            : ((size_t)(bb * NHd + hh) * Seq + ss) * HDm + dd;
        dst[idx] = (unsigned short)bfbits(val);
      }
    }
  }
}

// ---------------------------------------------------------------------------
// Flash attention with TDM-staged K/V tiles.
// grid (B*NH, Seq/128); 8 waves/block, 16 q-rows per wave.
// Per kv tile of 32: wave0 DMAs K(32x64) and Vt(64x32) into a double-buffered
// LDS stage via tensor_load_to_lds; all waves read WMMA fragments from LDS.
// Q,K: [bh][s][d] bf16;  Vt: [bh][d][s] bf16;  out: [b][s][H] f32
// ---------------------------------------------------------------------------
__global__ __launch_bounds__(256) void attention_kernel(
    const unsigned short* __restrict__ Q,
    const unsigned short* __restrict__ K,
    const unsigned short* __restrict__ Vt,
    float* __restrict__ out) {
  __shared__ unsigned short kbuf[2][32 * HDm];   // [kv row][d]
  __shared__ unsigned short vbuf[2][HDm * 32];   // [d][kv]
  __shared__ unsigned short plds[8 * 16 * 32];   // per-wave P re-layout slab

  const int wave = threadIdx.x >> 5;
  const int lane = threadIdx.x & 31;
  const int nn   = lane & 15;
  const int half = lane >> 4;
  const int bh   = blockIdx.x;
  const int b    = bh / NHd;
  const int h    = bh % NHd;
  const int q0   = blockIdx.y * 128 + wave * 16;

  const unsigned short* Qh = Q  + (size_t)bh * Seq * HDm;
  const unsigned short* Kh = K  + (size_t)bh * Seq * HDm;
  const unsigned short* Vh = Vt + (size_t)bh * HDm * Seq;
  unsigned short* pl = plds + wave * (16 * 32);

  // Prime the pipeline: tile 0 into buffer 0.
  if (wave == 0) {
    tdm_load_2d(lds_off(kbuf[0]), Kh, HDm, 32, HDm, Seq, HDm);
    tdm_load_2d(lds_off(vbuf[0]), Vh, 32, HDm, Seq, HDm, Seq);
  }

  const v16bf aq0 = load_fragA(Qh + (size_t)q0 * HDm,      HDm);
  const v16bf aq1 = load_fragA(Qh + (size_t)q0 * HDm + 32, HDm);

  float mrow[8], lrow[8];
#pragma unroll
  for (int r = 0; r < 8; r++) { mrow[r] = -3.0e38f; lrow[r] = 0.0f; }
  v8f acc0 = {}, acc1 = {}, acc2 = {}, acc3 = {};

  const float SC = 0.125f;   // 1/sqrt(HD)
  const int ntiles = Seq / 32;

  for (int jt = 0; jt < ntiles; ++jt) {
    const int buf = jt & 1;
    if (wave == 0) {
      __builtin_amdgcn_s_wait_tensorcnt(0);      // tile `buf` landed in LDS
    }
    __syncthreads();   // tile ready; everyone also done with buffer buf^1
    if (wave == 0 && jt + 1 < ntiles) {
      const int j2 = (jt + 1) * 32;
      tdm_load_2d(lds_off(kbuf[buf ^ 1]), Kh + (size_t)j2 * HDm,
                  HDm, 32, HDm, Seq, HDm);
      tdm_load_2d(lds_off(vbuf[buf ^ 1]), Vh + j2,
                  32, HDm, Seq, HDm, Seq);
    }

    const unsigned short* kb = kbuf[buf];
    const unsigned short* vb = vbuf[buf];

    // S = Q @ K^T : 16x32 score tile (two 16x16 N-tiles, K-dim 64)
    v8f s0 = {}, s1 = {};
    {
      v16bf bk;
      bk = load_fragB(kb,                    HDm); s0 = wmma_bf16(aq0, bk, s0);
      bk = load_fragB(kb + 32,               HDm); s0 = wmma_bf16(aq1, bk, s0);
      bk = load_fragB(kb + 16 * HDm,         HDm); s1 = wmma_bf16(aq0, bk, s1);
      bk = load_fragB(kb + 16 * HDm + 32,    HDm); s1 = wmma_bf16(aq1, bk, s1);
    }

    // Online softmax; each 16-lane half owns rows {half*8 + r}
#pragma unroll
    for (int r = 0; r < 8; r++) {
      float v0 = s0[r] * SC;
      float v1 = s1[r] * SC;
      float tm = fmaxf(v0, v1);
      tm = fmaxf(tm, __shfl_xor(tm, 1, 16));
      tm = fmaxf(tm, __shfl_xor(tm, 2, 16));
      tm = fmaxf(tm, __shfl_xor(tm, 4, 16));
      tm = fmaxf(tm, __shfl_xor(tm, 8, 16));
      const float mn   = fmaxf(mrow[r], tm);
      const float corr = __expf(mrow[r] - mn);
      mrow[r] = mn;
      const float p0 = __expf(v0 - mn);
      const float p1 = __expf(v1 - mn);
      float rs = p0 + p1;
      rs += __shfl_xor(rs, 1, 16);
      rs += __shfl_xor(rs, 2, 16);
      rs += __shfl_xor(rs, 4, 16);
      rs += __shfl_xor(rs, 8, 16);
      lrow[r] = lrow[r] * corr + rs;
      acc0[r] *= corr; acc1[r] *= corr; acc2[r] *= corr; acc3[r] *= corr;
      const int prow = half * 8 + r;
      pl[prow * 32 + nn]      = (unsigned short)bfbits(p0);
      pl[prow * 32 + 16 + nn] = (unsigned short)bfbits(p1);
    }

    // ctx += P(16x32) @ V(32x64): P back as A-fragment from LDS
    const v16bf ap = load_fragA(pl, 32);
    v16bf bv;
    bv = load_fragB(vb + 0  * 32, 32); acc0 = wmma_bf16(ap, bv, acc0);
    bv = load_fragB(vb + 16 * 32, 32); acc1 = wmma_bf16(ap, bv, acc1);
    bv = load_fragB(vb + 32 * 32, 32); acc2 = wmma_bf16(ap, bv, acc2);
    bv = load_fragB(vb + 48 * 32, 32); acc3 = wmma_bf16(ap, bv, acc3);
  }

#pragma unroll
  for (int r = 0; r < 8; r++) {
    const float inv = 1.0f / lrow[r];
    const int tok = q0 + half * 8 + r;
    float* orow = out + ((size_t)b * Seq + tok) * Hid + h * HDm + nn;
    orow[0]  = acc0[r] * inv;
    orow[16] = acc1[r] * inv;
    orow[32] = acc2[r] * inv;
    orow[48] = acc3[r] * inv;
  }
}

// ---------------------------------------------------------------------------
extern "C" void kernel_launch(void* const* d_in, const int* in_sizes, int n_in,
                              void* d_out, int out_size, void* d_ws, size_t ws_size,
                              hipStream_t stream) {
  const float* hidden = (const float*)d_in[0];
  const float* Wq = (const float*)d_in[1];
  const float* bq = (const float*)d_in[2];
  const float* Wk = (const float*)d_in[3];
  const float* bk = (const float*)d_in[4];
  const float* Wv = (const float*)d_in[5];
  const float* bv = (const float*)d_in[6];
  float* out = (float*)d_out;

  // workspace layout (bf16 / u16 units), ~40 MB total
  unsigned short* ws = (unsigned short*)d_ws;
  size_t o = 0;
  unsigned short* Xb  = ws + o; o += (size_t)Mtok * Hid;  // hidden bf16
  unsigned short* Wqb = ws + o; o += (size_t)Hid * Hid;
  unsigned short* Wkb = ws + o; o += (size_t)Hid * Hid;
  unsigned short* Wvb = ws + o; o += (size_t)Hid * Hid;
  unsigned short* Qb  = ws + o; o += (size_t)Mtok * Hid;  // [b,h,s,d]
  unsigned short* Kb  = ws + o; o += (size_t)Mtok * Hid;  // [b,h,s,d]
  unsigned short* Vtb = ws + o;                           // [b,h,d,s]

  {
    int n4 = Mtok * Hid / 4;
    f32_to_bf16_kernel<<<n4 / 256, 256, 0, stream>>>(hidden, Xb, n4);
    n4 = Hid * Hid / 4;
    f32_to_bf16_kernel<<<n4 / 256, 256, 0, stream>>>(Wq, Wqb, n4);
    f32_to_bf16_kernel<<<n4 / 256, 256, 0, stream>>>(Wk, Wkb, n4);
    f32_to_bf16_kernel<<<n4 / 256, 256, 0, stream>>>(Wv, Wvb, n4);
  }

  dim3 ggrid(Mtok / 64, Hid / 128);
  qkv_gemm_kernel<<<ggrid, 256, 0, stream>>>(Xb, Wqb, bq, Qb, 0);
  qkv_gemm_kernel<<<ggrid, 256, 0, stream>>>(Xb, Wkb, bk, Kb, 0);
  qkv_gemm_kernel<<<ggrid, 256, 0, stream>>>(Xb, Wvb, bv, Vtb, 1);

  dim3 agrid(Bsz * NHd, Seq / 128);
  attention_kernel<<<agrid, 256, 0, stream>>>(Qb, Kb, Vtb, out);
}
